// RbG_framework_73349451481856
// MI455X (gfx1250) — matrix operand order repaired and checked by reading, not compiled, not run on hardware
//
#include <hip/hip_runtime.h>
#include <hip/hip_bf16.h>
#include <math.h>

typedef __attribute__((ext_vector_type(16))) _Float16 v16h;
typedef __attribute__((ext_vector_type(8)))  _Float16 v8h;
typedef __attribute__((ext_vector_type(4)))  _Float16 v4h;
typedef __attribute__((ext_vector_type(8)))  float    v8f;
typedef __attribute__((ext_vector_type(4)))  unsigned int v4u;
typedef __attribute__((ext_vector_type(8)))  int      v8i;
typedef __attribute__((ext_vector_type(4)))  int      v4i;

#define NB    2
#define CCH   128
#define HIMG  96
#define WIMG  96
#define HWPX  (HIMG * WIMG)          // 9216
#define NPIX  (NB * HWPX)            // 18432
#define HID   256
#define P2    25
#define LRELU_SLOPE 0.2f
#define GN_EPS 1e-5f

#if defined(__has_builtin)
#if __has_builtin(__builtin_amdgcn_tensor_load_to_lds)
#define HAVE_TDM 1
#endif
#endif

// ---------------------------------------------------------------------------
// Plain f32 -> f16 convert (weights)
// ---------------------------------------------------------------------------
__global__ __launch_bounds__(256) void k_cvt_plain(const float* __restrict__ src,
                                                   _Float16* __restrict__ dst,
                                                   int count) {
  int i = blockIdx.x * 256 + threadIdx.x;
  if (i < count) dst[i] = (_Float16)src[i];
}

// ---------------------------------------------------------------------------
// [n][c][hw] f32 -> [n*hw][c] f16 (channels innermost for WMMA B fragments)
// ---------------------------------------------------------------------------
__global__ __launch_bounds__(256) void k_cvt_tr(const float* __restrict__ src,
                                                _Float16* __restrict__ dst) {
  int i = blockIdx.x * 256 + threadIdx.x;
  if (i >= NB * CCH * HWPX) return;
  int n = i / (CCH * HWPX);
  int r = i % (CCH * HWPX);
  int c = r / HWPX;
  int s = r % HWPX;
  dst[((size_t)n * HWPX + s) * CCH + c] = (_Float16)src[i];
}

// ---------------------------------------------------------------------------
// WMMA GEMM: Y[pix][M] = Wh[M][K] * Xh[pix][K]^T
// Block = 4 waves; block tile = 64(M) x 64(pix); wave tile = 16(M) x 64(pix).
// A macro-tile (64 x K f16) staged in LDS via the Tensor Data Mover once per
// block; each wave then reuses one A fragment across 4 WMMAs (4 pixel tiles).
// Fragment k order: contiguous k = 16*(lane/16) + s in both A and B (see round-1 comments).
// Optional epilogues: bias, LeakyReLU, residual add, secondary f16 output.
// ---------------------------------------------------------------------------
__global__ __launch_bounds__(128)
void k_gemm_wmma(const _Float16* __restrict__ Wh,   // [M][K]
                 const _Float16* __restrict__ Xh,   // [NPIX][K]
                 float* __restrict__ Y,             // [NPIX][M] or null
                 _Float16* __restrict__ Yh,         // [NPIX][M] f16 or null
                 const float* __restrict__ bias,    // [M] or null
                 const float* __restrict__ resid,   // [NPIX][M] or null
                 int M, int K, int lrelu) {
  __shared__ _Float16 sA[64 * 256];                 // 32 KB (K <= 256)
  const int wave  = threadIdx.x >> 5;
  const int lane  = threadIdx.x & 31;
  const int mbase = blockIdx.y * 64;
  const int row   = lane & 15;
  const int hi    = lane >> 4;

  // ---- stage A tile [64][K] into LDS ----
#if defined(HAVE_TDM)
  if (wave == 0) {
    unsigned long long ga = (unsigned long long)(const void*)(Wh + (size_t)mbase * K);
    unsigned lds_off = (unsigned)(size_t)(void*)sA;
    // D# group0: count=1 | lds_addr | global_addr[56:0] | type=2
    v4u g0 = (v4u){1u, lds_off, (unsigned)ga,
                   ((unsigned)(ga >> 32) & 0x01ffffffu) | (2u << 30)};
    // D# group1: data_size=2B, tensor [K x M], tile [K x 64], stride K
    int d0 = (1 << 16);                                   // data_size = 1 (2B)
    int d1 = (K & 0xffff) << 16;                          // tensor_dim0 lo
    int d2 = ((K >> 16) & 0xffff) | ((M & 0xffff) << 16); // dim0 hi | dim1 lo
    int d3 = ((M >> 16) & 0xffff) | ((K & 0xffff) << 16); // dim1 hi | tile_dim0
    int d4 = 64;                                          // tile_dim1 = 64 rows
    int d5 = K;                                           // tensor_dim0_stride lo
    v8i g1 = (v8i){d0, d1, d2, d3, d4, d5, 0, 0};
    v4i gz = (v4i){0, 0, 0, 0};
#if __clang_major__ >= 23
    v8i gz8 = (v8i){0, 0, 0, 0, 0, 0, 0, 0};
    __builtin_amdgcn_tensor_load_to_lds(g0, g1, gz, gz, gz8, 0);
#else
    __builtin_amdgcn_tensor_load_to_lds(g0, g1, gz, gz, 0);
#endif
    __builtin_amdgcn_s_wait_tensorcnt(0);
  }
  __syncthreads();
#else
  for (int j = threadIdx.x; j < (64 * K) / 8; j += 128) {
    int r = j / (K / 8), c = (j % (K / 8)) * 8;
    *(v8h*)&sA[r * K + c] = *(const v8h*)(Wh + (size_t)(mbase + r) * K + c);
  }
  __syncthreads();
#endif

  // ---- per-wave fragment base pointers ----
  const _Float16* arow = &sA[(wave * 16 + row) * K + hi * 16];
  const _Float16* b0 = Xh + (size_t)(blockIdx.x * 64 +  0 + row) * K + hi * 16;
  const _Float16* b1 = Xh + (size_t)(blockIdx.x * 64 + 16 + row) * K + hi * 16;
  const _Float16* b2 = Xh + (size_t)(blockIdx.x * 64 + 32 + row) * K + hi * 16;
  const _Float16* b3 = Xh + (size_t)(blockIdx.x * 64 + 48 + row) * K + hi * 16;

  v8f acc0 = {}, acc1 = {}, acc2 = {}, acc3 = {};
  for (int kk = 0; kk < K; kk += 32) {
    v8h a_lo = *(const v8h*)(arow + kk);
    v8h a_hi = *(const v8h*)(arow + kk + 8);
    v16h a;
#pragma unroll
    for (int i = 0; i < 8; ++i) { a[i] = a_lo[i]; a[i + 8] = a_hi[i]; }
    if (kk + 32 < K) {                         // gfx1250 global_prefetch_b8
      __builtin_prefetch(b0 + kk + 32, 0, 1);
      __builtin_prefetch(b1 + kk + 32, 0, 1);
      __builtin_prefetch(b2 + kk + 32, 0, 1);
      __builtin_prefetch(b3 + kk + 32, 0, 1);
    }
    v16h f0 = *(const v16h*)(b0 + kk);
    v16h f1 = *(const v16h*)(b1 + kk);
    v16h f2 = *(const v16h*)(b2 + kk);
    v16h f3 = *(const v16h*)(b3 + kk);
    acc0 = __builtin_amdgcn_wmma_f32_16x16x32_f16(false, a, false, f0, (short)0, acc0, false, false);
    acc1 = __builtin_amdgcn_wmma_f32_16x16x32_f16(false, a, false, f1, (short)0, acc1, false, false);
    acc2 = __builtin_amdgcn_wmma_f32_16x16x32_f16(false, a, false, f2, (short)0, acc2, false, false);
    acc3 = __builtin_amdgcn_wmma_f32_16x16x32_f16(false, a, false, f3, (short)0, acc3, false, false);
  }

  // ---- epilogue ----
  v8f* accs[4] = {&acc0, &acc1, &acc2, &acc3};
#pragma unroll
  for (int s2 = 0; s2 < 4; ++s2) {
    int pixg = blockIdx.x * 64 + s2 * 16 + row;
    v8f a = *accs[s2];
#pragma unroll
    for (int r = 0; r < 8; ++r) {
      int mg  = mbase + wave * 16 + r + hi * 8;
      float y = a[r];
      if (bias)  y += bias[mg];
      if (lrelu) y  = y >= 0.f ? y : LRELU_SLOPE * y;
      if (resid) y += resid[(size_t)pixg * M + mg];
      if (Y)  Y [(size_t)pixg * M + mg] = y;
      if (Yh) Yh[(size_t)pixg * M + mg] = (_Float16)y;
    }
  }
}

// ---------------------------------------------------------------------------
// Deformable 25-tap attention. One wave32 per pixel.
// lane = head*4 + part ; lane owns channels [head*16 + part*4, +4)
// Output written directly in f16 [pix][C] (feeds the fc WMMA GEMM).
// ---------------------------------------------------------------------------
__device__ __forceinline__ float bilin_dot(const float* __restrict__ f, int n,
                                           float sx, float sy, int ch0,
                                           float4 q4) {
  float x0f = floorf(sx), y0f = floorf(sy);
  int x0 = (int)x0f, y0 = (int)y0f;
  float wx1 = sx - x0f, wx0 = 1.f - wx1;
  float wy1 = sy - y0f, wy0 = 1.f - wy1;
  float r = 0.f;
#pragma unroll
  for (int cy = 0; cy < 2; ++cy)
#pragma unroll
    for (int cx = 0; cx < 2; ++cx) {
      int xi = x0 + cx, yi = y0 + cy;
      if (xi >= 0 && xi < WIMG && yi >= 0 && yi < HIMG) {
        float wgt = (cx ? wx1 : wx0) * (cy ? wy1 : wy0);
        const float4 g = *(const float4*)(f + ((size_t)n * HWPX + yi * WIMG + xi) * CCH + ch0);
        r += wgt * (g.x * q4.x + g.y * q4.y + g.z * q4.z + g.w * q4.w);
      }
    }
  return r;
}

__device__ __forceinline__ float4 bilin_vec(const float* __restrict__ f, int n,
                                            float sx, float sy, int ch0) {
  float x0f = floorf(sx), y0f = floorf(sy);
  int x0 = (int)x0f, y0 = (int)y0f;
  float wx1 = sx - x0f, wx0 = 1.f - wx1;
  float wy1 = sy - y0f, wy0 = 1.f - wy1;
  float4 r = make_float4(0.f, 0.f, 0.f, 0.f);
#pragma unroll
  for (int cy = 0; cy < 2; ++cy)
#pragma unroll
    for (int cx = 0; cx < 2; ++cx) {
      int xi = x0 + cx, yi = y0 + cy;
      if (xi >= 0 && xi < WIMG && yi >= 0 && yi < HIMG) {
        float wgt = (cx ? wx1 : wx0) * (cy ? wy1 : wy0);
        const float4 g = *(const float4*)(f + ((size_t)n * HWPX + yi * WIMG + xi) * CCH + ch0);
        r.x += wgt * g.x; r.y += wgt * g.y; r.z += wgt * g.z; r.w += wgt * g.w;
      }
    }
  return r;
}

__global__ __launch_bounds__(128)
void k_attn(const float* __restrict__ q,    // [NPIX][C]
            const float* __restrict__ k,    // [NPIX][C]
            const float* __restrict__ v,    // [NPIX][C]
            const float* __restrict__ df,   // [NB][2][HWPX]
            _Float16* __restrict__ out) {   // [NPIX][C] f16
  int pix  = blockIdx.x * 4 + (threadIdx.x >> 5);
  int lane = threadIdx.x & 31;
  if (pix >= NPIX) return;
  int n = pix / HWPX, s = pix % HWPX;
  int py = s / WIMG, px = s % WIMG;
  int head = lane >> 2, part = lane & 3;
  int ch0 = head * 16 + part * 4;

  const float4 q4 = *(const float4*)(q + (size_t)pix * CCH + ch0);
  float dx = df[(size_t)n * 2 * HWPX + s];
  float dy = df[(size_t)n * 2 * HWPX + HWPX + s];

  float l[P2];
  const float scale = 0.25f;                // 1/sqrt(d_k=16)
#pragma unroll
  for (int p = 0; p < P2; ++p) {
    float sx = (float)px + (float)(p % 5) - 2.f + dx;
    float sy = (float)py + (float)(p / 5) - 2.f + dy;
    float a = bilin_dot(k, n, sx, sy, ch0, q4);
    a += __shfl_xor(a, 1, 32);              // reduce across head's 4 lanes
    a += __shfl_xor(a, 2, 32);
    l[p] = a * scale;
  }

  float mx = l[0];
#pragma unroll
  for (int p = 1; p < P2; ++p) mx = fmaxf(mx, l[p]);
  float sum = 0.f;
#pragma unroll
  for (int p = 0; p < P2; ++p) { l[p] = __expf(l[p] - mx); sum += l[p]; }
  float inv = 1.f / sum;

  float4 o = make_float4(0.f, 0.f, 0.f, 0.f);
#pragma unroll
  for (int p = 0; p < P2; ++p) {
    float w = l[p] * inv;
    float sx = (float)px + (float)(p % 5) - 2.f + dx;
    float sy = (float)py + (float)(p / 5) - 2.f + dy;
    float4 vs = bilin_vec(v, n, sx, sy, ch0);
    o.x += w * vs.x; o.y += w * vs.y; o.z += w * vs.z; o.w += w * vs.w;
  }
  v4h oh = (v4h){(_Float16)o.x, (_Float16)o.y, (_Float16)o.z, (_Float16)o.w};
  *(v4h*)(out + (size_t)pix * CCH + ch0) = oh;
}

// ---------------------------------------------------------------------------
// GroupNorm(1, C): two-pass.
// ---------------------------------------------------------------------------
__global__ void k_zero_stats(float* s) { if (threadIdx.x < 4) s[threadIdx.x] = 0.f; }

__global__ __launch_bounds__(256)
void k_gn_partial(const float* __restrict__ fin, float* __restrict__ stats) {
  __shared__ float s_sum[256];
  __shared__ float s_sq[256];
  int n = blockIdx.y;
  size_t base = (size_t)n * (CCH * HWPX) + (size_t)blockIdx.x * 1024 + threadIdx.x * 4;
  float4 x = *(const float4*)(fin + base);
  s_sum[threadIdx.x] = x.x + x.y + x.z + x.w;
  s_sq[threadIdx.x]  = x.x * x.x + x.y * x.y + x.z * x.z + x.w * x.w;
  __syncthreads();
  for (int st = 128; st > 0; st >>= 1) {
    if (threadIdx.x < st) {
      s_sum[threadIdx.x] += s_sum[threadIdx.x + st];
      s_sq[threadIdx.x]  += s_sq[threadIdx.x + st];
    }
    __syncthreads();
  }
  if (threadIdx.x == 0) {
    atomicAdd(&stats[n * 2 + 0], s_sum[0]);
    atomicAdd(&stats[n * 2 + 1], s_sq[0]);
  }
}

__global__ __launch_bounds__(256)
void k_gn_finalize(const float* __restrict__ fin, const float* __restrict__ stats,
                   const float* __restrict__ gamma, const float* __restrict__ beta,
                   float* __restrict__ out) {
  int i = blockIdx.x * 256 + threadIdx.x;
  if (i >= NB * CCH * HWPX) return;
  int n = i / (CCH * HWPX);
  int r = i % (CCH * HWPX);
  int c = r / HWPX;
  int s = r % HWPX;
  const float cnt  = (float)(CCH * HWPX);
  float mean = stats[n * 2 + 0] / cnt;
  float var  = stats[n * 2 + 1] / cnt - mean * mean;
  float inv  = rsqrtf(var + GN_EPS);
  float val  = fin[((size_t)n * HWPX + s) * CCH + c];
  out[i] = (val - mean) * inv * gamma[c] + beta[c];
}

// ---------------------------------------------------------------------------
extern "C" void kernel_launch(void* const* d_in, const int* in_sizes, int n_in,
                              void* d_out, int out_size, void* d_ws, size_t ws_size,
                              hipStream_t stream) {
  (void)in_sizes; (void)n_in; (void)out_size; (void)ws_size;
  const float* query = (const float*)d_in[0];
  const float* key   = (const float*)d_in[1];
  const float* value = (const float*)d_in[2];
  const float* df    = (const float*)d_in[3];
  const float* w_q   = (const float*)d_in[4];
  const float* w_k   = (const float*)d_in[5];
  const float* w_v   = (const float*)d_in[6];
  const float* w_fc  = (const float*)d_in[7];
  const float* w1    = (const float*)d_in[8];
  const float* b1    = (const float*)d_in[9];
  const float* w2    = (const float*)d_in[10];
  const float* b2    = (const float*)d_in[11];
  const float* gam   = (const float*)d_in[12];
  const float* bet   = (const float*)d_in[13];
  float* out = (float*)d_out;

  // workspace layout (bytes)
  char* ws = (char*)d_ws;
  const size_t HXB = (size_t)NPIX * CCH * 2;   // f16 activation plane
  const size_t F32 = (size_t)NPIX * CCH * 4;   // f32 activation plane
  const size_t HX_OFF   = 262144;              // after f16 weights
  const size_t Q_OFF    = HX_OFF + HXB;
  const size_t K_OFF    = Q_OFF  + F32;
  const size_t V_OFF    = K_OFF  + F32;
  const size_t FC_OFF   = V_OFF  + F32;
  const size_t FIN_OFF  = FC_OFF + F32;
  const size_t STAT_OFF = FIN_OFF + F32;

  _Float16* wq_h  = (_Float16*)ws;
  _Float16* wk_h  = wq_h + 16384;
  _Float16* wv_h  = wq_h + 32768;
  _Float16* wfc_h = wq_h + 49152;
  _Float16* w1_h  = wq_h + 65536;              // [256][128]
  _Float16* w2_h  = wq_h + 98304;              // [128][256]
  _Float16* hx    = (_Float16*)(ws + HX_OFF);  // attn f16 out / gemm input
  float* qf    = (float*)(ws + Q_OFF);
  float* kf    = (float*)(ws + K_OFF);
  float* vf    = (float*)(ws + V_OFF);
  float* fc    = (float*)(ws + FC_OFF);
  float* fin   = (float*)(ws + FIN_OFF);
  float* stats = (float*)(ws + STAT_OFF);
  _Float16* hx2   = (_Float16*)(ws + V_OFF);   // alias: fc f16 (v dead)
  _Float16* hid_h = (_Float16*)(ws + K_OFF);   // alias: [NPIX][256] f16 (k dead)

  const int CVT_TOTAL = NB * CCH * HWPX;
  const dim3 gG128(NPIX / 64, 2);              // M=128
  const dim3 gG256(NPIX / 64, 4);              // M=256

  k_zero_stats<<<1, 32, 0, stream>>>(stats);

  // weight conversions
  k_cvt_plain<<<64, 256, 0, stream>>>(w_q,  wq_h, 16384);
  k_cvt_plain<<<64, 256, 0, stream>>>(w_k,  wk_h, 16384);
  k_cvt_plain<<<64, 256, 0, stream>>>(w_v,  wv_h, 16384);
  k_cvt_plain<<<64, 256, 0, stream>>>(w_fc, wfc_h, 16384);
  k_cvt_plain<<<128, 256, 0, stream>>>(w1,  w1_h, 32768);
  k_cvt_plain<<<128, 256, 0, stream>>>(w2,  w2_h, 32768);

  // q/k/v projections (f16 WMMA, f32 accumulate)
  k_cvt_tr<<<CVT_TOTAL / 256, 256, 0, stream>>>(query, hx);
  k_gemm_wmma<<<gG128, 128, 0, stream>>>(wq_h, hx, qf, nullptr, nullptr, nullptr, CCH, CCH, 0);
  k_cvt_tr<<<CVT_TOTAL / 256, 256, 0, stream>>>(key, hx);
  k_gemm_wmma<<<gG128, 128, 0, stream>>>(wk_h, hx, kf, nullptr, nullptr, nullptr, CCH, CCH, 0);
  k_cvt_tr<<<CVT_TOTAL / 256, 256, 0, stream>>>(value, hx);
  k_gemm_wmma<<<gG128, 128, 0, stream>>>(wv_h, hx, vf, nullptr, nullptr, nullptr, CCH, CCH, 0);

  // deformable attention -> f16 directly
  k_attn<<<NPIX / 4, 128, 0, stream>>>(qf, kf, vf, df, hx);

  // fc projection: f32 (residual base) + fused f16 copy for mlp1 input
  k_gemm_wmma<<<gG128, 128, 0, stream>>>(wfc_h, hx, fc, hx2, nullptr, nullptr, CCH, CCH, 0);

  // mlp1: bias + LeakyReLU, f16-only output (feeds mlp2)
  k_gemm_wmma<<<gG256, 128, 0, stream>>>(w1_h, hx2, nullptr, hid_h, b1, nullptr, HID, CCH, 1);

  // mlp2: bias + residual(fc) -> fin f32
  k_gemm_wmma<<<gG128, 128, 0, stream>>>(w2_h, hid_h, fin, nullptr, b2, fc, CCH, HID, 0);

  // GroupNorm
  k_gn_partial<<<dim3((CCH * HWPX) / 1024, NB), 256, 0, stream>>>(fin, stats);
  k_gn_finalize<<<(CVT_TOTAL + 255) / 256, 256, 0, stream>>>(fin, stats, gam, bet, out);
}
